// NeuralODE_75058848465133
// MI455X (gfx1250) — compile-verified, matrix operand order checked by reading, not run on hardware
//
#include <hip/hip_runtime.h>
#include <hip/hip_bf16.h>

// ---------------------------------------------------------------------------
// NeuralODE on MI455X (gfx1250): 100 sequential 1024x1024x1024 GEMMs via
// v_wmma_f32_16x16x32_bf16, f32 state + f32 accumulation, bf16 operands.
// ---------------------------------------------------------------------------

typedef __attribute__((ext_vector_type(16))) __bf16 v16bf;
typedef __attribute__((ext_vector_type(8)))  float  v8f;

#define DIM 1024
#define BM  64     // block tile M
#define BN  128    // block tile N
#define WM  32     // wave tile M
#define WN  32     // wave tile N
#define DT_STEP 0.1f

// ---- W (f32) -> Wb (bf16), elementwise, n is a multiple of 1024 ------------
__global__ __launch_bounds__(256)
void convert_w_kernel(const float* __restrict__ W, __bf16* __restrict__ Wb) {
    size_t i = ((size_t)blockIdx.x * blockDim.x + threadIdx.x) * 4;
    float4 v = *(const float4*)(W + i);
    Wb[i + 0] = (__bf16)v.x;
    Wb[i + 1] = (__bf16)v.y;
    Wb[i + 2] = (__bf16)v.z;
    Wb[i + 3] = (__bf16)v.w;
}

// ---- x -> traj[0] (f32 copy) and h_bf16 ------------------------------------
__global__ __launch_bounds__(256)
void init_h_kernel(const float* __restrict__ x, float* __restrict__ traj0,
                   __bf16* __restrict__ hb) {
    size_t i = ((size_t)blockIdx.x * blockDim.x + threadIdx.x) * 4;
    float4 v = *(const float4*)(x + i);
    *(float4*)(traj0 + i) = v;
    hb[i + 0] = (__bf16)v.x;
    hb[i + 1] = (__bf16)v.y;
    hb[i + 2] = (__bf16)v.z;
    hb[i + 3] = (__bf16)v.w;
}

// ---- features = traj[100] ---------------------------------------------------
__global__ __launch_bounds__(256)
void copy_out_kernel(const float* __restrict__ src, float* __restrict__ dst) {
    size_t i = ((size_t)blockIdx.x * blockDim.x + threadIdx.x) * 4;
    *(float4*)(dst + i) = *(const float4*)(src + i);
}

// ---- one Euler step: traj_out = traj_in + DT*tanh(h @ W^T + b) -------------
// hb_in : bf16 copy of current h        [DIM][DIM]
// Wb    : bf16 copy of W (slab, row n holds W[n][k] over k)  [DIM][DIM]
// bias  : f32 bias for this slab        [DIM]
// traj_in/traj_out : f32 state slabs in d_out
// hb_out: bf16 copy of next h (for the next step's A operand)
__global__ __launch_bounds__(256)
void ode_step_kernel(const __bf16* __restrict__ hb_in,
                     const __bf16* __restrict__ Wb,
                     const float*  __restrict__ bias,
                     const float*  __restrict__ traj_in,
                     float*        __restrict__ traj_out,
                     __bf16*       __restrict__ hb_out) {
    const int lane  = threadIdx.x & 31;
    const int wave  = threadIdx.x >> 5;         // 0..7
    const int waveM = wave >> 2;                // 0..1
    const int waveN = wave & 3;                 // 0..3
    const int blockM = blockIdx.y * BM;
    const int blockN = blockIdx.x * BN;

    const int l16 = lane & 15;                  // row (A) / col (B) within tile
    const int lhK = (lane >> 4) << 4;           // 0 or 16: K-half per half-wave

    const int mBase = blockM + waveM * WM;      // wave's 32 output rows
    const int nBase = blockN + waveN * WN;      // wave's 32 output cols

    // A fragment rows: contiguous K order (B uses the same), lane l%16 holds row m=l%16,
    // lanes 0-15 carry K=kc..kc+15, lanes 16-31 carry K=kc+16..kc+31.
    const __bf16* aPtr0 = hb_in + (size_t)(mBase + l16)      * DIM + lhK;
    const __bf16* aPtr1 = hb_in + (size_t)(mBase + 16 + l16) * DIM + lhK;
    // B fragment: lane l%16 holds column n=l%16 => 16 consecutive K elements
    // of original-W row n (B[k][n] = W[n][k]) -- contiguous, no transpose.
    const __bf16* bPtr0 = Wb + (size_t)(nBase + l16)      * DIM + lhK;
    const __bf16* bPtr1 = Wb + (size_t)(nBase + 16 + l16) * DIM + lhK;

    v8f c00 = {}, c01 = {}, c10 = {}, c11 = {};

    #pragma unroll 4
    for (int kc = 0; kc < DIM; kc += 32) {
        v16bf a0 = *(const v16bf*)(aPtr0 + kc);
        v16bf a1 = *(const v16bf*)(aPtr1 + kc);
        v16bf b0 = *(const v16bf*)(bPtr0 + kc);
        v16bf b1 = *(const v16bf*)(bPtr1 + kc);
        c00 = __builtin_amdgcn_wmma_f32_16x16x32_bf16(
                  false, a0, false, b0, (short)0, c00, false, false);
        c01 = __builtin_amdgcn_wmma_f32_16x16x32_bf16(
                  false, a0, false, b1, (short)0, c01, false, false);
        c10 = __builtin_amdgcn_wmma_f32_16x16x32_bf16(
                  false, a1, false, b0, (short)0, c10, false, false);
        c11 = __builtin_amdgcn_wmma_f32_16x16x32_bf16(
                  false, a1, false, b1, (short)0, c11, false, false);
    }

    // Epilogue. C/D layout: VGPR r, lanes 0-15 -> (m = r,   n = lane),
    //                               lanes 16-31 -> (m = r+8, n = lane-16).
    const int mOff = (lane >> 4) << 3;          // 0 or 8

    #pragma unroll
    for (int t = 0; t < 4; ++t) {
        const v8f c = (t == 0) ? c00 : (t == 1) ? c01 : (t == 2) ? c10 : c11;
        const int tm = mBase + ((t >> 1) << 4);         // +0 / +16
        const int n  = nBase + ((t & 1) << 4) + l16;    // +0 / +16
        const float bn = bias[n];
        #pragma unroll
        for (int r = 0; r < 8; ++r) {
            const int m = tm + mOff + r;
            const size_t idx = (size_t)m * DIM + n;
            const float f  = tanhf(c[r] + bn);
            const float hn = fmaf(DT_STEP, f, traj_in[idx]);
            traj_out[idx] = hn;
            hb_out[idx]   = (__bf16)hn;
        }
    }
}

extern "C" void kernel_launch(void* const* d_in, const int* in_sizes, int n_in,
                              void* d_out, int out_size, void* d_ws, size_t ws_size,
                              hipStream_t stream) {
    (void)in_sizes; (void)n_in; (void)out_size; (void)ws_size;

    const float* x = (const float*)d_in[0];                 // [1024,1024]
    const float* W = (const float*)d_in[1];                 // [10,1024,1024]
    const float* b = (const float*)d_in[2];                 // [10,1024]

    const size_t ONE_M = (size_t)DIM * DIM;                 // 1M elements

    float* features = (float*)d_out;                        // [1024,1024]
    float* traj     = (float*)d_out + ONE_M;                // [101,1024,1024]

    // workspace: Wb (20 MB bf16) + two ping-pong bf16 h buffers (2 MB each)
    __bf16* Wb  = (__bf16*)d_ws;
    __bf16* hb0 = (__bf16*)((char*)d_ws + (size_t)20 * 1024 * 1024);
    __bf16* hb1 = (__bf16*)((char*)d_ws + (size_t)22 * 1024 * 1024);

    // 1) W -> bf16 (10M elems, 4 per thread)
    convert_w_kernel<<<(10 * ONE_M) / (256 * 4), 256, 0, stream>>>(W, Wb);

    // 2) traj[0] = x, hb0 = bf16(x)
    init_h_kernel<<<ONE_M / (256 * 4), 256, 0, stream>>>(x, traj, hb0);

    // 3) 100 sequential Euler steps
    dim3 grid(DIM / BN, DIM / BM);                          // (8, 16) = 128 WGs
    for (int s = 0; s < 100; ++s) {
        const int slab = s / 10;                            // W/b index
        ode_step_kernel<<<grid, 256, 0, stream>>>(
            (s & 1) ? hb1 : hb0,
            Wb + (size_t)slab * ONE_M,
            b  + (size_t)slab * DIM,
            traj + (size_t)s * ONE_M,
            traj + (size_t)(s + 1) * ONE_M,
            (s & 1) ? hb0 : hb1);
    }

    // 4) features = traj[100]
    copy_out_kernel<<<ONE_M / (256 * 4), 256, 0, stream>>>(
        traj + (size_t)100 * ONE_M, features);
}
